// DCRS_32504312496806
// MI455X (gfx1250) — compile-verified, matrix operand order checked by reading, not run on hardware
//
#include <hip/hip_runtime.h>
#include <hip/hip_bf16.h>

// ---------------------------------------------------------------------------
// DCRS GNN for gfx1250 (MI455X). Dense 50000x128x128 GEMMs via
// v_wmma_f32_16x16x32_bf16 with pre-staged bf16 activations/weights so the
// GEMM inner loop is pure 16B loads + WMMA. Edge gather / edge-softmax /
// scatter via coalesced float4 + global_atomic_add_f32.
// ---------------------------------------------------------------------------

typedef __attribute__((ext_vector_type(16))) __bf16 v16bf;
typedef __attribute__((ext_vector_type(8)))  __bf16 v8bf;
typedef __attribute__((ext_vector_type(8)))  float  v8f;

#define NNODES 50000
#define NEDGES 800000
#define FEAT   128
#define SLOPE  0.2f

// ---- helpers --------------------------------------------------------------

static __device__ __forceinline__ __bf16 f2bf(float f) {
    return (__bf16)f;   // native f32->bf16 (RNE) on gfx1250
}

// monotonic f32 -> u32 encoding for atomicMax-based segment max
static __device__ __forceinline__ unsigned fenc(float x) {
    unsigned u = __float_as_uint(x);
    return (u & 0x80000000u) ? ~u : (u | 0x80000000u);
}
static __device__ __forceinline__ float fdec(unsigned u) {
    return (u & 0x80000000u) ? __uint_as_float(u ^ 0x80000000u)
                             : __uint_as_float(~u);
}

static __device__ __forceinline__ float wave_sum(float v) {
    #pragma unroll
    for (int o = 16; o > 0; o >>= 1) v += __shfl_down(v, o, 32);
    return v;
}

// ---- activation convert: X[N,128] f32 -> bf16, 8 elements/thread ----------

__global__ __launch_bounds__(256) void k_conv_x(const float* __restrict__ X,
                                                __bf16* __restrict__ Xb, int n8) {
    int i = blockIdx.x * 256 + threadIdx.x;          // one per 8 elements
    if (i >= n8) return;
    const float4 a = *(const float4*)(X + i * 8);
    const float4 b = *(const float4*)(X + i * 8 + 4);
    v8bf o;
    o[0] = f2bf(a.x); o[1] = f2bf(a.y); o[2] = f2bf(a.z); o[3] = f2bf(a.w);
    o[4] = f2bf(b.x); o[5] = f2bf(b.y); o[6] = f2bf(b.z); o[7] = f2bf(b.w);
    *(v8bf*)(Xb + i * 8) = o;
}

// ---- weight convert: Bt[n*128+k] = W[n*ldn + k*ldk] as bf16 ---------------

__global__ void k_conv_w(const float* __restrict__ W, __bf16* __restrict__ Bt,
                         int ldn, int ldk) {
    int idx = blockIdx.x * 256 + threadIdx.x;
    if (idx >= FEAT * FEAT) return;
    int n = idx >> 7, k = idx & 127;
    Bt[idx] = f2bf(W[n * ldn + k * ldk]);
}

// ---- WMMA GEMM: C[M,128] = A[M,128](bf16) * Bt[n][k](bf16) (+bias)(+relu) -
// one wave -> one 16x16 tile; 8 waves/block -> 16 rows x 128 cols.
// Inner loop: two 16B A loads + two 16B B loads + one v_wmma per K-step.

__global__ __launch_bounds__(256) void k_gemm_bf16(
    const __bf16* __restrict__ A, const __bf16* __restrict__ Bt,
    float* __restrict__ C, const float* __restrict__ bias, int doRelu)
{
    const int wave = threadIdx.x >> 5;
    const int lane = threadIdx.x & 31;
    const int half = lane >> 4;
    const int m0   = blockIdx.x * 16;
    const int mr   = m0 + (lane & 15);          // A row for this lane
    const int nc   = wave * 16 + (lane & 15);   // B/C column for this lane

    v8f c = {};
    #pragma unroll
    for (int ks = 0; ks < 4; ++ks) {
        const int kb  = ks * 32 + half * 8;     // elements 0..7  : K = kb..kb+7
        const int kb2 = kb + 16;                // elements 8..15 : K = kb2..kb2+7

        const v8bf a0 = *(const v8bf*)(A + mr * FEAT + kb);
        const v8bf a1 = *(const v8bf*)(A + mr * FEAT + kb2);
        const v8bf b0 = *(const v8bf*)(Bt + nc * FEAT + kb);
        const v8bf b1 = *(const v8bf*)(Bt + nc * FEAT + kb2);
        v16bf a, b;
        #pragma unroll
        for (int i = 0; i < 8; ++i) {
            a[i] = a0[i]; a[8 + i] = a1[i];
            b[i] = b0[i]; b[8 + i] = b1[i];
        }
        c = __builtin_amdgcn_wmma_f32_16x16x32_bf16(
                false, a, false, b, (short)0, c, false, false);
    }

    const float bv = bias ? bias[nc] : 0.0f;
    #pragma unroll
    for (int r = 0; r < 8; ++r) {
        float v = c[r] + bv;
        if (doRelu) v = fmaxf(v, 0.0f);
        C[(m0 + half * 8 + r) * FEAT + nc] = v;    // lanes 0-15: rows m0..m0+7
    }
}

// ---- GDN edge attention logits (width-128 hidden) -------------------------

__global__ void k_edge_logit128(const float* __restrict__ hs,
                                const float* __restrict__ ht,
                                const float* __restrict__ Wa,
                                const int* __restrict__ src,
                                const int* __restrict__ dst,
                                float* __restrict__ eOut, int E)
{
    const int edge = blockIdx.x * 8 + (threadIdx.x >> 5);
    const int lane = threadIdx.x & 31;
    if (edge >= E) return;
    const int s = src[edge], t = dst[edge];
    const float4 a = *(const float4*)(hs + s * FEAT + lane * 4);
    const float4 b = *(const float4*)(ht + t * FEAT + lane * 4);
    const float4 w = *(const float4*)(Wa + lane * 4);
    float z0 = a.x + b.x, z1 = a.y + b.y, z2 = a.z + b.z, z3 = a.w + b.w;
    z0 = z0 > 0.f ? z0 : SLOPE * z0;
    z1 = z1 > 0.f ? z1 : SLOPE * z1;
    z2 = z2 > 0.f ? z2 : SLOPE * z2;
    z3 = z3 > 0.f ? z3 : SLOPE * z3;
    float p = z0 * w.x + z1 * w.y + z2 * w.z + z3 * w.w;
    p = wave_sum(p);
    if (lane == 0) eOut[edge] = p;
}

// ---- GDN layer-3 edge logit (hidden dim 1) --------------------------------

__global__ void k_edge_logit1(const float* __restrict__ hs,
                              const float* __restrict__ ht,
                              const float* __restrict__ wa,
                              const int* __restrict__ src,
                              const int* __restrict__ dst,
                              float* __restrict__ eOut, int E)
{
    int i = blockIdx.x * blockDim.x + threadIdx.x;
    if (i >= E) return;
    float z = hs[src[i]] + ht[dst[i]];
    z = z > 0.f ? z : SLOPE * z;
    eOut[i] = wa[0] * z;
}

// ---- edge softmax (segmented by src) --------------------------------------

__global__ void k_fill_u32(unsigned* p, unsigned v, int n) {
    int i = blockIdx.x * blockDim.x + threadIdx.x;
    if (i < n) p[i] = v;
}
__global__ void k_fill_f32(float* p, float v, int n) {
    int i = blockIdx.x * blockDim.x + threadIdx.x;
    if (i < n) p[i] = v;
}
__global__ void k_fill_from(float* p, const float* src0, int n) {
    int i = blockIdx.x * blockDim.x + threadIdx.x;
    if (i < n) p[i] = src0[0];
}

__global__ void k_seg_max(const float* __restrict__ e, const int* __restrict__ src,
                          unsigned* __restrict__ m, int E) {
    int i = blockIdx.x * blockDim.x + threadIdx.x;
    if (i >= E) return;
    atomicMax(&m[src[i]], fenc(e[i]));
}

__global__ void k_exp_sum(float* __restrict__ e, const unsigned* __restrict__ m,
                          float* __restrict__ s, const int* __restrict__ src, int E) {
    int i = blockIdx.x * blockDim.x + threadIdx.x;
    if (i >= E) return;
    const int sn = src[i];
    const float ex = __expf(e[i] - fdec(m[sn]));
    e[i] = ex;                                  // overwrite logits with exp()
    atomicAdd(&s[sn], ex);
}

// ---- GDN message scatter: agg[dst] += h[src] * alpha ----------------------

__global__ void k_scatter_h(const float* __restrict__ h, const float* __restrict__ ex,
                            const float* __restrict__ s, const int* __restrict__ src,
                            const int* __restrict__ dst, float* __restrict__ agg, int E)
{
    const int edge = blockIdx.x * 8 + (threadIdx.x >> 5);
    const int lane = threadIdx.x & 31;
    if (edge >= E) return;
    const int sn = src[edge], dn = dst[edge];
    const float alpha = ex[edge] / s[sn];
    const float4 hv = *(const float4*)(h + sn * FEAT + lane * 4);
    float* o = agg + dn * FEAT + lane * 4;
    atomicAdd(o + 0, hv.x * alpha);
    atomicAdd(o + 1, hv.y * alpha);
    atomicAdd(o + 2, hv.z * alpha);
    atomicAdd(o + 3, hv.w * alpha);
}

// ---- GCN scatter: out[arow] += avals * support[acol] ----------------------

__global__ void k_gcn_scatter(const float* __restrict__ sup, const float* __restrict__ av,
                              const int* __restrict__ arow, const int* __restrict__ acol,
                              float* __restrict__ out, int E)
{
    const int edge = blockIdx.x * 8 + (threadIdx.x >> 5);
    const int lane = threadIdx.x & 31;
    if (edge >= E) return;
    const int rn = arow[edge], cn = acol[edge];
    const float w = av[edge];
    const float4 sv = *(const float4*)(sup + cn * FEAT + lane * 4);
    float* o = out + rn * FEAT + lane * 4;
    atomicAdd(o + 0, sv.x * w);
    atomicAdd(o + 1, sv.y * w);
    atomicAdd(o + 2, sv.z * w);
    atomicAdd(o + 3, sv.w * w);
}

__global__ void k_gcn_scatter1(const float* __restrict__ sup, const float* __restrict__ av,
                               const int* __restrict__ arow, const int* __restrict__ acol,
                               float* __restrict__ out, int E) {
    int i = blockIdx.x * blockDim.x + threadIdx.x;
    if (i >= E) return;
    atomicAdd(&out[arow[i]], av[i] * sup[acol[i]]);
}

// ---- GEMV: out[n] = dot(X[n,:128], w) + bias[0] ---------------------------

__global__ void k_gemv128(const float* __restrict__ X, const float* __restrict__ w,
                          const float* __restrict__ bias, float* __restrict__ out, int n)
{
    const int node = blockIdx.x * 8 + (threadIdx.x >> 5);
    const int lane = threadIdx.x & 31;
    if (node >= n) return;
    const float4 x  = *(const float4*)(X + node * FEAT + lane * 4);
    const float4 ww = *(const float4*)(w + lane * 4);
    float p = x.x * ww.x + x.y * ww.y + x.z * ww.z + x.w * ww.w;
    p = wave_sum(p);
    if (lane == 0) out[node] = p + (bias ? bias[0] : 0.0f);
}

// ---- elementwise ----------------------------------------------------------

__global__ void k_bias_bcast(float* __restrict__ out, const float* __restrict__ b, int n) {
    int i = blockIdx.x * blockDim.x + threadIdx.x;
    if (i < n) out[i] = b[i & 127];
}
__global__ void k_relu(float* p, int n) {
    int i = blockIdx.x * blockDim.x + threadIdx.x;
    if (i < n) p[i] = fmaxf(p[i], 0.0f);
}
__global__ void k_final(const float* __restrict__ diff, const float* __restrict__ role,
                        float* __restrict__ out, int n) {
    int i = blockIdx.x * blockDim.x + threadIdx.x;
    if (i >= n) return;
    const float x = 0.5f * (diff[i] + role[i]);     // (1-FUSE)*diff + FUSE*role, FUSE=0.5
    out[i] = 1.0f / (1.0f + __expf(-x));
}

// ---------------------------------------------------------------------------

extern "C" void kernel_launch(void* const* d_in, const int* in_sizes, int n_in,
                              void* d_out, int out_size, void* d_ws, size_t ws_size,
                              hipStream_t stream)
{
    (void)in_sizes; (void)n_in; (void)out_size; (void)ws_size;

    const int N = NNODES, E = NEDGES, F = FEAT;

    const float* feature = (const float*)d_in[0];
    const int*   src     = (const int*)d_in[1];
    const int*   dst     = (const int*)d_in[2];
    const int*   arow    = (const int*)d_in[3];
    const int*   acol    = (const int*)d_in[4];
    const float* avals   = (const float*)d_in[5];
    const float* P[28];
    for (int i = 0; i < 28; ++i) P[i] = (const float*)d_in[6 + i];
    // per layer l: P[7l]=Ws, +1=Wt, +2=Wa, +3=Wfc, +4=bfc, +5=gcnW, +6=gcnb

    float* ws = (float*)d_ws;
    float*    dbuf = ws;                     // [N,128] GDN features
    float*    bufA = ws + (size_t)N * F;     // [N,128]
    float*    bufB = ws + (size_t)2 * N * F; // [N,128]
    float*    bufC = ws + (size_t)3 * N * F; // [N,128]  agg / scatter target
    float*    eBuf = ws + (size_t)4 * N * F; // [E] logits -> exp
    float*    sBuf = eBuf + E;               // [N] softmax denom
    unsigned* mBuf = (unsigned*)(sBuf + N);  // [N] softmax max (encoded)
    float*    diff = (float*)(mBuf + N);     // [N]
    float*    role = diff + N;               // [N]
    float*    v1   = role + N;               // [N] hs3 / support3
    float*    v2   = v1 + N;                 // [N] ht3
    __bf16*   Wbf  = (__bf16*)(v2 + N);      // [128*128] staged bf16 weight
    __bf16*   Abf  = Wbf + (size_t)F * F;    // [N,128]   staged bf16 activations

    const dim3 B256(256);
    const dim3 gW((F * F + 255) / 256);      // 64
    const dim3 gGemm(N / 16);                // 3125
    const dim3 gConvX((N * F / 8 + 255) / 256); // 3125 (8 elems/thread)
    const dim3 gEdgeW(E / 8);                // 100000 (wave per edge)
    const dim3 gEdgeT((E + 255) / 256);      // 3125  (thread per edge)
    const dim3 gNodeW((N + 7) / 8);          // 6250  (wave per node)
    const dim3 gN((N + 255) / 256);
    const dim3 gNF((N * F + 255) / 256);     // 25000

    // ===================== GDN branch (diff score) =====================
    for (int l = 0; l < 3; ++l) {
        const float* h = (l == 0) ? feature : dbuf;
        // stage h to bf16 once; feeds both Ws and Wt GEMMs
        k_conv_x<<<gConvX, B256, 0, stream>>>(h, Abf, N * F / 8);
        // hs = h @ Ws.T   (Bt[n][k] = Ws[n][k] -> ldn=128, ldk=1)
        k_conv_w<<<gW, B256, 0, stream>>>(P[7*l + 0], Wbf, F, 1);
        k_gemm_bf16<<<gGemm, B256, 0, stream>>>(Abf, Wbf, bufA, nullptr, 0);
        // ht = h @ Wt.T
        k_conv_w<<<gW, B256, 0, stream>>>(P[7*l + 1], Wbf, F, 1);
        k_gemm_bf16<<<gGemm, B256, 0, stream>>>(Abf, Wbf, bufB, nullptr, 0);
        // e = Wa . leaky_relu(hs[src] + ht[dst])
        k_edge_logit128<<<gEdgeW, B256, 0, stream>>>(bufA, bufB, P[7*l + 2],
                                                     src, dst, eBuf, E);
        // edge softmax segmented by src
        k_fill_u32<<<gN, B256, 0, stream>>>(mBuf, 0u, N);
        k_seg_max<<<gEdgeT, B256, 0, stream>>>(eBuf, src, mBuf, E);
        k_fill_f32<<<gN, B256, 0, stream>>>(sBuf, 0.0f, N);
        k_exp_sum<<<gEdgeT, B256, 0, stream>>>(eBuf, mBuf, sBuf, src, E);
        // agg[dst] += h[src] * alpha
        k_fill_f32<<<gNF, B256, 0, stream>>>(bufC, 0.0f, N * F);
        k_scatter_h<<<gEdgeW, B256, 0, stream>>>(h, eBuf, sBuf, src, dst, bufC, E);
        // d = relu(agg @ Wfc.T + bfc)
        k_conv_x<<<gConvX, B256, 0, stream>>>(bufC, Abf, N * F / 8);
        k_conv_w<<<gW, B256, 0, stream>>>(P[7*l + 3], Wbf, F, 1);
        k_gemm_bf16<<<gGemm, B256, 0, stream>>>(Abf, Wbf, dbuf, P[7*l + 4], 1);
    }
    // ---- GDN layer 3 (out dim = 1) ----
    k_gemv128<<<gNodeW, B256, 0, stream>>>(dbuf, P[21], nullptr, v1, N);  // hs3
    k_gemv128<<<gNodeW, B256, 0, stream>>>(dbuf, P[22], nullptr, v2, N);  // ht3
    k_edge_logit1<<<gEdgeT, B256, 0, stream>>>(v1, v2, P[23], src, dst, eBuf, E);
    k_fill_u32<<<gN, B256, 0, stream>>>(mBuf, 0u, N);
    k_seg_max<<<gEdgeT, B256, 0, stream>>>(eBuf, src, mBuf, E);
    k_fill_f32<<<gN, B256, 0, stream>>>(sBuf, 0.0f, N);
    k_exp_sum<<<gEdgeT, B256, 0, stream>>>(eBuf, mBuf, sBuf, src, E);
    k_fill_f32<<<gNF, B256, 0, stream>>>(bufC, 0.0f, N * F);
    k_scatter_h<<<gEdgeW, B256, 0, stream>>>(dbuf, eBuf, sBuf, src, dst, bufC, E);
    k_gemv128<<<gNodeW, B256, 0, stream>>>(bufC, P[24], P[25], diff, N);  // Wfc3+bfc3

    // ===================== GCN branch (role score) =====================
    // x ping-pongs: feature -> bufA -> bufC -> bufA ; support in bufB
    const float* x = feature;
    float* outp = bufA;
    for (int l = 0; l < 3; ++l) {
        // support = x @ W   (W is [in,out]: Bt[n][k] = W[k][n] -> ldn=1, ldk=128)
        k_conv_x<<<gConvX, B256, 0, stream>>>(x, Abf, N * F / 8);
        k_conv_w<<<gW, B256, 0, stream>>>(P[7*l + 5], Wbf, 1, F);
        k_gemm_bf16<<<gGemm, B256, 0, stream>>>(Abf, Wbf, bufB, nullptr, 0);
        // out = b + segment_sum(avals * support[acol], arow); relu
        k_bias_bcast<<<gNF, B256, 0, stream>>>(outp, P[7*l + 6], N * F);
        k_gcn_scatter<<<gEdgeW, B256, 0, stream>>>(bufB, avals, arow, acol, outp, E);
        k_relu<<<gNF, B256, 0, stream>>>(outp, N * F);
        x = outp;
        outp = (outp == bufA) ? bufC : bufA;
    }
    // ---- GCN layer 3 (out dim = 1): W3 is [128,1] -> contiguous vector ----
    k_gemv128<<<gNodeW, B256, 0, stream>>>(x, P[26], nullptr, v1, N);     // support3
    k_fill_from<<<gN, B256, 0, stream>>>(role, P[27], N);                 // += b3
    k_gcn_scatter1<<<gEdgeT, B256, 0, stream>>>(v1, avals, arow, acol, role, E);

    // ===================== fuse + sigmoid =====================
    k_final<<<gN, B256, 0, stream>>>(diff, role, (float*)d_out, N);
}
